// Model_53815940219422
// MI455X (gfx1250) — matrix operand order repaired and checked by reading, not compiled, not run on hardware
//
#include <hip/hip_runtime.h>
#include <hip/hip_bf16.h>
#include <math.h>

// ---------------------------------------------------------------------------
// Types / WMMA helpers (CDNA5 / gfx1250, wave32)
// ---------------------------------------------------------------------------
typedef __attribute__((ext_vector_type(16))) __bf16 v16bf;
typedef __attribute__((ext_vector_type(2)))  __bf16 v2bf;
typedef __attribute__((ext_vector_type(8)))  float  v8f;

#define WMMA_BF16(a, b, c) \
  __builtin_amdgcn_wmma_f32_16x16x32_bf16(false, (a), false, (b), (short)0, (c), false, false)

// native f32->bf16 convert (RNE); compiler lowers to hardware cvt on gfx1250
static __device__ __forceinline__ __bf16 f2bf(float f) { return (__bf16)f; }

static __device__ __forceinline__ unsigned bfpack2(float a, float b) {
  v2bf p;
  p[0] = (__bf16)a;
  p[1] = (__bf16)b;
  return __builtin_bit_cast(unsigned, p);
}

// 16-bit A-matrix 16x32 fragment (ISA 7.12.2): per lane, elements 0..7 hold
// K=base..base+7 and elements 8..15 hold K=base+16..base+23.
static __device__ __forceinline__ v16bf load_a_frag(const __bf16* p) {
  v16bf r;
#pragma unroll
  for (int i = 0; i < 8; ++i) { r[i] = p[i]; r[i + 8] = p[i + 16]; }
  return r;
}

static __device__ __forceinline__ v8f vzero8() {
  v8f z = {0.f, 0.f, 0.f, 0.f, 0.f, 0.f, 0.f, 0.f};
  return z;
}

static __device__ __forceinline__ float gelu_f(float x) {
  return 0.5f * x * (1.f + erff(x * 0.70710678118654752f));
}

// Convert 16 consecutive floats (4x float4) into 2x uint4 of packed bf16.
static __device__ __forceinline__ void pack16(const float4* p, uint4& u0, uint4& u1) {
  float4 f0 = p[0], f1 = p[1], f2 = p[2], f3 = p[3];
  u0.x = bfpack2(f0.x, f0.y); u0.y = bfpack2(f0.z, f0.w);
  u0.z = bfpack2(f1.x, f1.y); u0.w = bfpack2(f1.z, f1.w);
  u1.x = bfpack2(f2.x, f2.y); u1.y = bfpack2(f2.z, f2.w);
  u1.z = bfpack2(f3.x, f3.y); u1.w = bfpack2(f3.z, f3.w);
}

// ---------------------------------------------------------------------------
// GEMM: C[M,N] = A[M,K] * W[N,K]^T + bias, optional exact GELU.
// 64x64 tile, BK=32, 128 threads (4 waves), each wave: 32x32 = 2x2 WMMA frags.
// K must be a multiple of 32 (call sites: 512/1536/2048). A,W 16B-aligned.
// ---------------------------------------------------------------------------
#define GROWP 48   // padded LDS row (bf16): 96B => every row 16B/32B aligned

__global__ __launch_bounds__(128) void gemm_bf16_kernel(
    const float* __restrict__ A, const float* __restrict__ W,
    const float* __restrict__ bias, float* __restrict__ C,
    int M, int N, int K, int act) {
  __shared__ __bf16 As[64 * GROWP];
  __shared__ __bf16 Bs[64 * GROWP];

  const int tid  = threadIdx.x;
  const int lane = tid & 31;
  const int wave = tid >> 5;
  const int wm = wave >> 1, wn = wave & 1;
  const int hi  = lane >> 4;       // lane half (0: lanes 0-15, 1: lanes 16-31)
  const int l16 = lane & 15;
  const int bm = blockIdx.x * 64, bn = blockIdx.y * 64;

  v8f acc[2][2];
#pragma unroll
  for (int mi = 0; mi < 2; ++mi)
#pragma unroll
    for (int ni = 0; ni < 2; ++ni) acc[mi][ni] = vzero8();

  const int lr  = tid >> 1;            // 0..63: tile row
  const int lc0 = (tid & 1) * 16;      // 0 or 16: 16-float chunk
  const bool okA = (bm + lr) < M;
  const bool okB = (bn + lr) < N;
  const float* apBase = A + (size_t)(bm + lr) * K + lc0;
  const float* wpBase = W + (size_t)(bn + lr) * K + lc0;
  __bf16* da = As + lr * GROWP + lc0;  // 16B aligned (96*lr + 2*lc0)
  __bf16* db = Bs + lr * GROWP + lc0;
  const uint4 z4 = {0u, 0u, 0u, 0u};

  for (int k0 = 0; k0 < K; k0 += 32) {
    uint4 a0, a1, b0, b1;
    if (okA) { pack16((const float4*)(apBase + k0), a0, a1); }
    else     { a0 = z4; a1 = z4; }
    if (okB) { pack16((const float4*)(wpBase + k0), b0, b1); }
    else     { b0 = z4; b1 = z4; }
    *(uint4*)(da)     = a0;
    *(uint4*)(da + 8) = a1;
    *(uint4*)(db)     = b0;
    *(uint4*)(db + 8) = b1;
    __syncthreads();

    v16bf afr[2], bfr[2];
#pragma unroll
    for (int mi = 0; mi < 2; ++mi) {
      const __bf16* p = As + (wm * 32 + mi * 16 + l16) * GROWP + hi * 8;
      afr[mi] = load_a_frag(p);
    }
#pragma unroll
    for (int ni = 0; ni < 2; ++ni) {
      const __bf16* p = Bs + (wn * 32 + ni * 16 + l16) * GROWP + hi * 8;
      bfr[ni] = load_a_frag(p);
    }
#pragma unroll
    for (int mi = 0; mi < 2; ++mi)
#pragma unroll
      for (int ni = 0; ni < 2; ++ni)
        acc[mi][ni] = WMMA_BF16(afr[mi], bfr[ni], acc[mi][ni]);
    __syncthreads();
  }

  // Epilogue. C/D layout: vgpr r -> row r (+8 for lane half 1), col = lane%16.
#pragma unroll
  for (int mi = 0; mi < 2; ++mi)
#pragma unroll
    for (int ni = 0; ni < 2; ++ni)
#pragma unroll
      for (int r = 0; r < 8; ++r) {
        const int row = bm + wm * 32 + mi * 16 + r + hi * 8;
        const int col = bn + wn * 32 + ni * 16 + l16;
        if (row < M && col < N) {
          float v = acc[mi][ni][r];
          if (bias) v += bias[col];
          if (act == 1) v = gelu_f(v);
          C[(size_t)row * N + col] = v;
        }
      }
}

// ---------------------------------------------------------------------------
// Flash attention (WMMA).  Q,K,V,O live in merged [B, L, H*64] f32 layout.
// grid = (Lq/64, B*H); block = 128 (4 waves); each wave owns 16 query rows.
// K tile staged [k][d], V staged transposed [d][k] in LDS so both B-fragments
// are contiguous v16bf loads.  P goes through per-wave LDS (C->A relayout).
// ---------------------------------------------------------------------------
#define FROWP 80   // padded LDS row (bf16): 160B => 32B aligned rows

__global__ __launch_bounds__(128) void flash_attn_kernel(
    const float* __restrict__ Q, const float* __restrict__ Kp,
    const float* __restrict__ Vp, float* __restrict__ O,
    int Lq, int Lk, int causal) {
  __shared__ __bf16 Ks[64 * FROWP];
  __shared__ __bf16 Vt[64 * FROWP];
  __shared__ __bf16 Ps[4][16 * FROWP];

  const int tid  = threadIdx.x;
  const int lane = tid & 31;
  const int wave = tid >> 5;
  const int hi = lane >> 4, l16 = lane & 15;
  const int b = blockIdx.y >> 3;
  const int h = blockIdx.y & 7;
  const int q0 = blockIdx.x * 64;
  const int qw = q0 + wave * 16;
  const float scale = 0.125f;  // 1/sqrt(64), folded into Q

  // Q A-fragments (two 32-wide K halves over d = 0..63), scaled.
  v16bf aq[2];
  {
    const float* qp = Q + ((size_t)(b * Lq + qw + l16)) * 512 + h * 64;
#pragma unroll
    for (int half = 0; half < 2; ++half) {
      const int base = half * 32 + hi * 8;
      float4 lo0 = *(const float4*)(qp + base);
      float4 lo1 = *(const float4*)(qp + base + 4);
      float4 hi0 = *(const float4*)(qp + base + 16);
      float4 hi1 = *(const float4*)(qp + base + 20);
      v16bf a;
      a[0] = f2bf(lo0.x * scale); a[1] = f2bf(lo0.y * scale);
      a[2] = f2bf(lo0.z * scale); a[3] = f2bf(lo0.w * scale);
      a[4] = f2bf(lo1.x * scale); a[5] = f2bf(lo1.y * scale);
      a[6] = f2bf(lo1.z * scale); a[7] = f2bf(lo1.w * scale);
      a[8]  = f2bf(hi0.x * scale); a[9]  = f2bf(hi0.y * scale);
      a[10] = f2bf(hi0.z * scale); a[11] = f2bf(hi0.w * scale);
      a[12] = f2bf(hi1.x * scale); a[13] = f2bf(hi1.y * scale);
      a[14] = f2bf(hi1.z * scale); a[15] = f2bf(hi1.w * scale);
      aq[half] = a;
    }
  }

  v8f oacc[4];
#pragma unroll
  for (int od = 0; od < 4; ++od) oacc[od] = vzero8();
  float mrow[8], lrow[8];
#pragma unroll
  for (int r = 0; r < 8; ++r) { mrow[r] = -1e30f; lrow[r] = 0.f; }

  const int nTiles = causal ? ((q0 + 63) / 64 + 1) : (Lk / 64);

  // cooperative-load mapping: each thread owns 2 k-rows x 16 d
  const int kr = (tid & 31) * 2;       // 0,2,..,62
  const int d0 = (tid >> 5) * 16;      // 0,16,32,48

  for (int kt = 0; kt < nTiles; ++kt) {
    const int kbase = kt * 64;
    {
      const size_t g0 = ((size_t)(b * Lk + kbase + kr)) * 512 + h * 64 + d0;
      uint4 ku0a, ku0b, ku1a, ku1b;
      pack16((const float4*)(Kp + g0), ku0a, ku0b);
      pack16((const float4*)(Kp + g0 + 512), ku1a, ku1b);
      __bf16* kd0 = Ks + kr * FROWP + d0;          // 16B aligned
      __bf16* kd1 = Ks + (kr + 1) * FROWP + d0;
      *(uint4*)(kd0)     = ku0a;  *(uint4*)(kd0 + 8) = ku0b;
      *(uint4*)(kd1)     = ku1a;  *(uint4*)(kd1 + 8) = ku1b;

      float v0[16], v1[16];
      *(float4*)&v0[0]  = *(const float4*)(Vp + g0);
      *(float4*)&v0[4]  = *(const float4*)(Vp + g0 + 4);
      *(float4*)&v0[8]  = *(const float4*)(Vp + g0 + 8);
      *(float4*)&v0[12] = *(const float4*)(Vp + g0 + 12);
      *(float4*)&v1[0]  = *(const float4*)(Vp + g0 + 512);
      *(float4*)&v1[4]  = *(const float4*)(Vp + g0 + 516);
      *(float4*)&v1[8]  = *(const float4*)(Vp + g0 + 520);
      *(float4*)&v1[12] = *(const float4*)(Vp + g0 + 524);
#pragma unroll
      for (int j = 0; j < 16; ++j)     // transposed: columns kr,kr+1 packed b32
        *(unsigned*)(Vt + (d0 + j) * FROWP + kr) = bfpack2(v0[j], v1[j]);
    }
    __syncthreads();

    // S = (Q*scale) . K^T   (4 subtiles of 16 cols, 2 WMMA each over d)
    v8f s[4];
#pragma unroll
    for (int ni = 0; ni < 4; ++ni) {
      v8f sa = vzero8();
#pragma unroll
      for (int half = 0; half < 2; ++half) {
        const __bf16* bp = Ks + (ni * 16 + l16) * FROWP + half * 32 + hi * 8;
        v16bf bf = load_a_frag(bp);
        sa = WMMA_BF16(aq[half], bf, sa);
      }
      s[ni] = sa;
    }

    // Online softmax update, one row per (r, lane-half).
#pragma unroll
    for (int r = 0; r < 8; ++r) {
      const int qg = qw + r + hi * 8;
      float tmax = -1e30f;
#pragma unroll
      for (int ni = 0; ni < 4; ++ni) {
        if (causal) {
          const int kg = kbase + ni * 16 + l16;
          if (kg > qg) s[ni][r] = -1e30f;
        }
        tmax = fmaxf(tmax, s[ni][r]);
      }
      for (int msk = 1; msk < 16; msk <<= 1)
        tmax = fmaxf(tmax, __shfl_xor(tmax, msk, 32));
      const float mn = fmaxf(mrow[r], tmax);
      const float corr = __expf(mrow[r] - mn);
#pragma unroll
      for (int od = 0; od < 4; ++od) oacc[od][r] *= corr;
      float psum = 0.f;
#pragma unroll
      for (int ni = 0; ni < 4; ++ni) {
        const float p = __expf(s[ni][r] - mn);
        psum += p;
        Ps[wave][(r + hi * 8) * FROWP + ni * 16 + l16] = f2bf(p);
      }
      for (int msk = 1; msk < 16; msk <<= 1)
        psum += __shfl_xor(psum, msk, 32);
      lrow[r] = lrow[r] * corr + psum;
      mrow[r] = mn;
    }

    // Wave-local LDS ordering before re-reading P as A-fragments.
    asm volatile("s_wait_dscnt 0" ::: "memory");

    // O += P . V
#pragma unroll
    for (int od = 0; od < 4; ++od) {
#pragma unroll
      for (int half = 0; half < 2; ++half) {
        const __bf16* pp = Ps[wave] + l16 * FROWP + half * 32 + hi * 8;
        v16bf ap = load_a_frag(pp);
        const __bf16* vb = Vt + (od * 16 + l16) * FROWP + half * 32 + hi * 8;
        v16bf bv = load_a_frag(vb);
        oacc[od] = WMMA_BF16(ap, bv, oacc[od]);
      }
    }
    __syncthreads();
  }

#pragma unroll
  for (int od = 0; od < 4; ++od)
#pragma unroll
    for (int r = 0; r < 8; ++r) {
      const int row = qw + r + hi * 8;
      const float v = oacc[od][r] / lrow[r];
      O[((size_t)(b * Lq + row)) * 512 + h * 64 + od * 16 + l16] = v;
    }
}

// ---------------------------------------------------------------------------
// Elementwise / small kernels
// ---------------------------------------------------------------------------
#define PE_NLF 0.01798894601f  // ln(10000)/512

__global__ void embed_enc_kernel(const float* __restrict__ x,
                                 const float* __restrict__ w,
                                 float* __restrict__ out) {
  const int idx = blockIdx.x * blockDim.x + threadIdx.x;
  const int total = 4 * 2048 * 512;
  if (idx >= total) return;
  const int dm = idx & 511;
  const int l  = (idx >> 9) & 2047;
  const int b  = idx >> 20;
  float acc = 0.f;
#pragma unroll
  for (int t = 0; t < 3; ++t) {
    const int ls = (l + t - 1 + 2048) & 2047;
    const float* xp = x + ((size_t)(b * 2048 + ls)) * 7;
#pragma unroll
    for (int c = 0; c < 7; ++c) acc += xp[c] * w[(dm * 7 + c) * 3 + t];
  }
  const float freq = __expf(-(float)(dm & ~1) * PE_NLF);
  const float ang = (float)l * freq;
  acc += (dm & 1) ? __cosf(ang) : __sinf(ang);
  out[idx] = acc;
}

__global__ void embed_dec_kernel(const float* __restrict__ x_enc,
                                 const float* __restrict__ w,
                                 float* __restrict__ out) {
  const int idx = blockIdx.x * blockDim.x + threadIdx.x;
  const int total = 4 * 512 * 512;
  if (idx >= total) return;
  const int dm = idx & 511;
  const int l  = (idx >> 9) & 511;
  const int b  = idx >> 18;
  float acc = 0.f;
#pragma unroll
  for (int t = 0; t < 3; ++t) {
    const int ls = (l + t - 1 + 512) & 511;
    if (ls < 256) {  // x_dec = [x_enc[:, -256:, :], zeros]
      const float* xp = x_enc + ((size_t)(b * 2048 + 1792 + ls)) * 7;
#pragma unroll
      for (int c = 0; c < 7; ++c) acc += xp[c] * w[(dm * 7 + c) * 3 + t];
    }
  }
  const float freq = __expf(-(float)(dm & ~1) * PE_NLF);
  const float ang = (float)l * freq;
  acc += (dm & 1) ? __cosf(ang) : __sinf(ang);
  out[idx] = acc;
}

// out = LayerNorm(x + res) — one wave per 512-wide row
__global__ __launch_bounds__(256) void ln_res_kernel(
    const float* __restrict__ x, const float* __restrict__ res,
    const float* __restrict__ g, const float* __restrict__ bv,
    float* __restrict__ out, int M) {
  const int lane = threadIdx.x & 31;
  const int row = blockIdx.x * 8 + (threadIdx.x >> 5);
  if (row >= M) return;
  const size_t base = (size_t)row * 512;
  float v[16];
  float sum = 0.f;
#pragma unroll
  for (int j = 0; j < 4; ++j) {
    const int c = (lane + j * 32) * 4;
    float4 e = *(const float4*)(x + base + c);
    if (res) {
      float4 rr = *(const float4*)(res + base + c);
      e.x += rr.x; e.y += rr.y; e.z += rr.z; e.w += rr.w;
    }
    v[j * 4 + 0] = e.x; v[j * 4 + 1] = e.y; v[j * 4 + 2] = e.z; v[j * 4 + 3] = e.w;
    sum += e.x + e.y + e.z + e.w;
  }
  for (int m = 1; m < 32; m <<= 1) sum += __shfl_xor(sum, m, 32);
  const float mean = sum * (1.f / 512.f);
  float sq = 0.f;
#pragma unroll
  for (int j = 0; j < 16; ++j) { const float d = v[j] - mean; sq += d * d; }
  for (int m = 1; m < 32; m <<= 1) sq += __shfl_xor(sq, m, 32);
  const float rstd = rsqrtf(sq * (1.f / 512.f) + 1e-5f);
#pragma unroll
  for (int j = 0; j < 4; ++j) {
    const int c = (lane + j * 32) * 4;
    float4 gg = *(const float4*)(g + c);
    float4 bb = *(const float4*)(bv + c);
    float4 o;
    o.x = (v[j * 4 + 0] - mean) * rstd * gg.x + bb.x;
    o.y = (v[j * 4 + 1] - mean) * rstd * gg.y + bb.y;
    o.z = (v[j * 4 + 2] - mean) * rstd * gg.z + bb.z;
    o.w = (v[j * 4 + 3] - mean) * rstd * gg.w + bb.w;
    *(float4*)(out + base + c) = o;
  }
}

__global__ void im2col_kernel(const float* __restrict__ h, float* __restrict__ x3) {
  const int idx = blockIdx.x * blockDim.x + threadIdx.x;
  const int total = 4 * 2048 * 1536;
  if (idx >= total) return;
  const int j = idx % 1536;
  const int l = (idx / 1536) & 2047;
  const int b = idx / (1536 * 2048);
  const int c = j / 3, t = j % 3;
  const int ls = (l + t - 1 + 2048) & 2047;
  x3[idx] = h[((size_t)(b * 2048 + ls)) * 512 + c];
}

__global__ void bn_elu_kernel(float* __restrict__ y, const float* __restrict__ g,
                              const float* __restrict__ b, int n) {
  const int idx = blockIdx.x * blockDim.x + threadIdx.x;
  if (idx >= n) return;
  const int c = idx & 511;
  const float v = y[idx] * (g[c] * rsqrtf(1.f + 1e-5f)) + b[c];
  y[idx] = v > 0.f ? v : (__expf(v) - 1.f);
}

__global__ void maxpool_kernel(const float* __restrict__ y, float* __restrict__ out) {
  const int idx = blockIdx.x * blockDim.x + threadIdx.x;
  const int total = 4 * 1024 * 512;
  if (idx >= total) return;
  const int d = idx & 511;
  const int lo = (idx >> 9) & 1023;
  const int b = idx >> 19;
  float m = -1e30f;
#pragma unroll
  for (int w = 0; w < 3; ++w) {
    const int li = 2 * lo - 1 + w;
    if (li >= 0 && li < 2048) m = fmaxf(m, y[((size_t)(b * 2048 + li)) * 512 + d]);
  }
  out[idx] = m;
}

__global__ void extract_kernel(const float* __restrict__ z, float* __restrict__ out) {
  const int idx = blockIdx.x * blockDim.x + threadIdx.x;
  const int total = 4 * 256 * 7;
  if (idx >= total) return;
  const int c = idx % 7;
  const int p = (idx / 7) & 255;
  const int b = idx / (7 * 256);
  out[idx] = z[((size_t)(b * 512 + 256 + p)) * 7 + c];
}

// ---------------------------------------------------------------------------
// Host
// ---------------------------------------------------------------------------
namespace {
struct AttnP { const float *qw, *qb, *kw, *kb, *vw, *vb, *ow, *ob; };
struct FfnP  { const float *w1, *b1, *w2, *b2; };
struct LnP   { const float *g, *b; };
}

extern "C" void kernel_launch(void* const* d_in, const int* in_sizes, int n_in,
                              void* d_out, int out_size, void* d_ws, size_t ws_size,
                              hipStream_t stream) {
  if (n_in < 71) return;
  auto F = [&](int i) -> const float* { return (const float*)d_in[i]; };

  const float *x_enc, *enc_emb, *dec_emb, *conv_w, *conv_b, *bng, *bnb, *proj_w, *proj_b;
  AttnP encA[2], dself, dcross;
  FfnP  encF[2], dffn;
  LnP   encN1[2], encN2[2], dn1, dn2, dn3, encNorm, decNorm;

  const bool insertion = (in_sizes[0] == 4 * 2048 * 7);
  if (insertion) {
    // insertion-order flattening: x_enc first, params in construction order
    int i = 0;
    x_enc = F(i++); enc_emb = F(i++); dec_emb = F(i++);
    for (int l = 0; l < 2; ++l) {
      encA[l].qw = F(i++); encA[l].qb = F(i++); encA[l].kw = F(i++); encA[l].kb = F(i++);
      encA[l].vw = F(i++); encA[l].vb = F(i++); encA[l].ow = F(i++); encA[l].ob = F(i++);
      encF[l].w1 = F(i++); encF[l].b1 = F(i++); encF[l].w2 = F(i++); encF[l].b2 = F(i++);
      encN1[l].g = F(i++); encN1[l].b = F(i++); encN2[l].g = F(i++); encN2[l].b = F(i++);
    }
    conv_w = F(i++); conv_b = F(i++); bng = F(i++); bnb = F(i++);
    encNorm.g = F(i++); encNorm.b = F(i++);
    dself.qw = F(i++); dself.qb = F(i++); dself.kw = F(i++); dself.kb = F(i++);
    dself.vw = F(i++); dself.vb = F(i++); dself.ow = F(i++); dself.ob = F(i++);
    dcross.qw = F(i++); dcross.qb = F(i++); dcross.kw = F(i++); dcross.kb = F(i++);
    dcross.vw = F(i++); dcross.vb = F(i++); dcross.ow = F(i++); dcross.ob = F(i++);
    dffn.w1 = F(i++); dffn.b1 = F(i++); dffn.w2 = F(i++); dffn.b2 = F(i++);
    dn1.g = F(i++); dn1.b = F(i++); dn2.g = F(i++); dn2.b = F(i++); dn3.g = F(i++); dn3.b = F(i++);
    decNorm.g = F(i++); decNorm.b = F(i++);
    proj_w = F(i++); proj_b = F(i++);
  } else {
    // JAX sorted-pytree flattening ('params' < 'x_enc', dict keys sorted)
    conv_b = F(0); bnb = F(1); bng = F(2); conv_w = F(3);
    dec_emb = F(4);
    dcross.kb = F(5);  dcross.kw = F(6);  dcross.ob = F(7);  dcross.ow = F(8);
    dcross.qb = F(9);  dcross.qw = F(10); dcross.vb = F(11); dcross.vw = F(12);
    dffn.b1 = F(13); dffn.b2 = F(14); dffn.w1 = F(15); dffn.w2 = F(16);
    dn1.b = F(17); dn1.g = F(18); dn2.b = F(19); dn2.g = F(20); dn3.b = F(21); dn3.g = F(22);
    dself.kb = F(23); dself.kw = F(24); dself.ob = F(25); dself.ow = F(26);
    dself.qb = F(27); dself.qw = F(28); dself.vb = F(29); dself.vw = F(30);
    decNorm.b = F(31); decNorm.g = F(32);
    enc_emb = F(33);
    for (int l = 0; l < 2; ++l) {
      const int s = 34 + l * 16;
      encA[l].kb = F(s + 0);  encA[l].kw = F(s + 1);  encA[l].ob = F(s + 2);  encA[l].ow = F(s + 3);
      encA[l].qb = F(s + 4);  encA[l].qw = F(s + 5);  encA[l].vb = F(s + 6);  encA[l].vw = F(s + 7);
      encF[l].b1 = F(s + 8);  encF[l].b2 = F(s + 9);  encF[l].w1 = F(s + 10); encF[l].w2 = F(s + 11);
      encN1[l].b = F(s + 12); encN1[l].g = F(s + 13); encN2[l].b = F(s + 14); encN2[l].g = F(s + 15);
    }
    encNorm.b = F(66); encNorm.g = F(67);
    proj_b = F(68); proj_w = F(69);
    x_enc = F(70);
  }

  // Workspace layout (floats)
  const size_t ACT = (size_t)4 * 2048 * 512;   // 4.19M floats
  const size_t BIG = (size_t)4 * 2048 * 2048;  // 16.78M floats
  const size_t needBytes = (6 * ACT + BIG) * sizeof(float);
  if (ws_size < needBytes) return;
  float* w = (float*)d_ws;
  size_t off = 0;
  auto alloc = [&](size_t n) { float* p = w + off; off += n; return p; };
  float* H  = alloc(ACT);   // encoder stream
  float* T  = alloc(ACT);   // temp / decoder stream
  float* Qb = alloc(ACT);
  float* Kb = alloc(ACT);
  float* Vb = alloc(ACT);
  float* Ab = alloc(ACT);   // attention out
  float* Fb = alloc(BIG);   // ffn hidden / im2col / o-proj temp

  auto gemm = [&](const float* A, const float* W, const float* bias, float* C,
                  int M, int N, int K, int act) {
    dim3 g((M + 63) / 64, (N + 63) / 64);
    gemm_bf16_kernel<<<g, 128, 0, stream>>>(A, W, bias, C, M, N, K, act);
  };
  auto lnres = [&](const float* x, const float* res, const LnP& n, float* out, int M) {
    ln_res_kernel<<<(M + 7) / 8, 256, 0, stream>>>(x, res, n.g, n.b, out, M);
  };
  auto blks = [](int n) { return (n + 255) / 256; };

  // ---------------- Encoder ----------------
  embed_enc_kernel<<<blks(4 * 2048 * 512), 256, 0, stream>>>(x_enc, enc_emb, H);

  auto enc_layer = [&](int l, int L) {
    const int M = 4 * L;
    gemm(H, encA[l].qw, encA[l].qb, Qb, M, 512, 512, 0);
    gemm(H, encA[l].kw, encA[l].kb, Kb, M, 512, 512, 0);
    gemm(H, encA[l].vw, encA[l].vb, Vb, M, 512, 512, 0);
    flash_attn_kernel<<<dim3(L / 64, 32), 128, 0, stream>>>(Qb, Kb, Vb, Ab, L, L, 0);
    gemm(Ab, encA[l].ow, encA[l].ob, T, M, 512, 512, 0);
    lnres(H, T, encN1[l], H, M);
    gemm(H, encF[l].w1, encF[l].b1, Fb, M, 2048, 512, 1);   // GELU fused
    gemm(Fb, encF[l].w2, encF[l].b2, T, M, 512, 2048, 0);
    lnres(H, T, encN2[l], H, M);
  };

  enc_layer(0, 2048);

  // conv distil: circular conv (as K=1536 GEMM) + BN-scale + ELU + maxpool/2
  im2col_kernel<<<blks(4 * 2048 * 1536), 256, 0, stream>>>(H, Fb);
  gemm(Fb, conv_w, conv_b, T, 4 * 2048, 512, 1536, 0);
  bn_elu_kernel<<<blks(4 * 2048 * 512), 256, 0, stream>>>(T, bng, bnb, 4 * 2048 * 512);
  maxpool_kernel<<<blks(4 * 1024 * 512), 256, 0, stream>>>(T, H);

  enc_layer(1, 1024);
  lnres(H, nullptr, encNorm, H, 4 * 1024);    // enc_norm; H = cross memory

  // ---------------- Decoder (L=512, M=2048) ----------------
  float* D = T;
  embed_dec_kernel<<<blks(4 * 512 * 512), 256, 0, stream>>>(x_enc, dec_emb, D);

  // self attention (causal)
  gemm(D, dself.qw, dself.qb, Qb, 2048, 512, 512, 0);
  gemm(D, dself.kw, dself.kb, Kb, 2048, 512, 512, 0);
  gemm(D, dself.vw, dself.vb, Vb, 2048, 512, 512, 0);
  flash_attn_kernel<<<dim3(8, 32), 128, 0, stream>>>(Qb, Kb, Vb, Ab, 512, 512, 1);
  gemm(Ab, dself.ow, dself.ob, Fb, 2048, 512, 512, 0);
  lnres(D, Fb, dn1, D, 2048);

  // cross attention against encoder memory (Lk = 1024)
  gemm(D, dcross.qw, dcross.qb, Qb, 2048, 512, 512, 0);
  gemm(H, dcross.kw, dcross.kb, Kb, 4096, 512, 512, 0);
  gemm(H, dcross.vw, dcross.vb, Vb, 4096, 512, 512, 0);
  flash_attn_kernel<<<dim3(8, 32), 128, 0, stream>>>(Qb, Kb, Vb, Ab, 512, 1024, 0);
  gemm(Ab, dcross.ow, dcross.ob, Fb, 2048, 512, 512, 0);
  lnres(D, Fb, dn2, D, 2048);

  // FFN
  gemm(D, dffn.w1, dffn.b1, Fb, 2048, 2048, 512, 1);
  gemm(Fb, dffn.w2, dffn.b2, Qb, 2048, 512, 2048, 0);
  lnres(D, Qb, dn3, D, 2048);
  lnres(D, nullptr, decNorm, D, 2048);

  // projection + slice out[:, -256:, :]
  gemm(D, proj_w, proj_b, Kb, 2048, 7, 512, 0);
  extract_kernel<<<blks(4 * 256 * 7), 256, 0, stream>>>(Kb, (float*)d_out);
}
